// FlashAttention_38336878084668
// MI455X (gfx1250) — matrix operand order repaired and checked by reading, not compiled, not run on hardware
//
#include <hip/hip_runtime.h>
#include <hip/hip_bf16.h>

typedef __attribute__((ext_vector_type(16))) _Float16 v16h;
typedef __attribute__((ext_vector_type(8)))  _Float16 v8h;
typedef __attribute__((ext_vector_type(8)))  float    v8f;

#define DIM    768
#define HEADS  12
#define HD     64
#define INNER  768
#define NQKV   2304
#define SEQ    2048
#define BATCH  2
#define ROWS   (BATCH * SEQ)   // 4096

// ---------- WMMA helpers ----------

__device__ __forceinline__ v8f wmma16(v16h a, v16h b, v8f c) {
  return __builtin_amdgcn_wmma_f32_16x16x32_f16(false, a, false, b, (short)0, c,
                                                false, false);
}

// A-matrix 16x32 f16 fragment from row-major LDS tile.
// Lane holds row m = lane&15; elements 0..7 = K (kbase + h*8 .. +7),
// elements 8..15 = K (kbase + 16 + h*8 .. +7), h = lane>>4.
__device__ __forceinline__ v16h afrag_lds(const _Float16* tile, int pitch,
                                          int m, int kbase, int lane) {
  const int h = (lane >> 4) & 1;
  const _Float16* p = tile + m * pitch + kbase + h * 8;
  v8h lo = *(const v8h*)(p);
  v8h hi = *(const v8h*)(p + 16);
  v16h f;
#pragma unroll
  for (int i = 0; i < 8; ++i) { f[i] = lo[i]; f[8 + i] = hi[i]; }
  return f;
}

// B-matrix 32x16 f16 fragment. LDS tile stores B transposed (row = n, col = k).
// Lane holds column n = lane&15; elements 0..7 = K (kbase + h*8 .. +7),
// elements 8..15 = K (kbase + 16 + h*8 .. +7), h = lane>>4.
__device__ __forceinline__ v16h bfrag_lds(const _Float16* tileT, int pitch,
                                          int n, int kbase, int lane) {
  const int h = (lane >> 4) & 1;
  const _Float16* p = tileT + n * pitch + kbase + h * 8;
  v8h lo = *(const v8h*)(p);
  v8h hi = *(const v8h*)(p + 16);
  v16h f;
#pragma unroll
  for (int i = 0; i < 8; ++i) { f[i] = lo[i]; f[8 + i] = hi[i]; }
  return f;
}

// ---------- Kernel 1: QKV projection  (x[4096x768] @ w[768x2304] + b) ----------
// Block tile 128(M) x 128(N), K-stage 64. 8 waves in a 4x2 grid; wave tile
// 32x64 => 2 A-frags x 4 B-frags -> 8 WMMAs per 32-K chunk.

__global__ __launch_bounds__(256) void qkv_kernel(
    const float* __restrict__ x, const float* __restrict__ w,
    const float* __restrict__ bias, _Float16* __restrict__ Q,
    _Float16* __restrict__ K, _Float16* __restrict__ V) {
  __shared__ __align__(16) _Float16 As[128 * 72];   // [m][k]
  __shared__ __align__(16) _Float16 Bs[128 * 72];   // transposed: [n][k]

  const int t = threadIdx.x;
  const int lane = t & 31;
  const int wv = t >> 5;
  const int wm = wv & 3;                  // M wave index (32-row stripe)
  const int wn = wv >> 2;                 // N wave index (64-col stripe)
  const int m0 = blockIdx.x * 128;
  const int n0 = blockIdx.y * 128;
  const int h = (lane >> 4) & 1;

  v8f acc[2][4] = {};
  for (int kc = 0; kc < DIM; kc += 64) {
    for (int i = t; i < 128 * 64; i += 256) {
      int r = i >> 6, c = i & 63;
      As[r * 72 + c] = (_Float16)x[(size_t)(m0 + r) * DIM + kc + c];
    }
    for (int i = t; i < 64 * 128; i += 256) {
      int k = i >> 7, n = i & 127;
      Bs[n * 72 + k] = (_Float16)w[(size_t)(kc + k) * NQKV + n0 + n];
    }
    if (kc + 64 < DIM) {  // prefetch next K chunk (global_prefetch_b8)
      __builtin_prefetch(&x[(size_t)(m0 + (t >> 1)) * DIM + kc + 64 + (t & 1) * 32], 0, 1);
      __builtin_prefetch(&w[(size_t)(kc + 64 + (t >> 2)) * NQKV + n0 + (t & 3) * 32], 0, 1);
    }
    __syncthreads();
#pragma unroll
    for (int kk = 0; kk < 2; ++kk) {
      v16h a0 = afrag_lds(As + (wm * 32) * 72, 72, lane & 15, kk * 32, lane);
      v16h a1 = afrag_lds(As + (wm * 32 + 16) * 72, 72, lane & 15, kk * 32, lane);
#pragma unroll
      for (int j = 0; j < 4; ++j) {
        v16h b = bfrag_lds(Bs + (wn * 64 + j * 16) * 72, 72, lane & 15,
                           kk * 32, lane);
        acc[0][j] = wmma16(a0, b, acc[0][j]);
        acc[1][j] = wmma16(a1, b, acc[1][j]);
      }
    }
    __syncthreads();
  }

#pragma unroll
  for (int ms = 0; ms < 2; ++ms)
#pragma unroll
    for (int j = 0; j < 4; ++j) {
      const int gn = n0 + wn * 64 + j * 16 + (lane & 15);
      const int ty = gn / INNER;               // 0=Q 1=K 2=V (uniform per tile)
      const int head = (gn % INNER) / HD;
      const int d = gn & (HD - 1);
      _Float16* dst = (ty == 0) ? Q : (ty == 1) ? K : V;
      const float bv = bias[gn];
#pragma unroll
      for (int r = 0; r < 8; ++r) {
        const int gm = m0 + wm * 32 + ms * 16 + r + 8 * h;
        const int b = gm >> 11, s = gm & (SEQ - 1);
        dst[(((size_t)b * HEADS + head) * SEQ + s) * HD + d] =
            (_Float16)(acc[ms][j][r] + bv);
      }
    }
}

// ---------- Kernel 2: flash attention ----------
// Grid: (S/64, B*HEADS). 128 threads = 4 waves; wave owns 16 query rows.
// KV tiles of 64 keys; online softmax in f32 registers; 16 WMMAs per tile.

__global__ __launch_bounds__(128) void attn_kernel(
    const _Float16* __restrict__ Q, const _Float16* __restrict__ Kg,
    const _Float16* __restrict__ V, _Float16* __restrict__ ctx) {
  __shared__ __align__(16) _Float16 Qs[64 * 72];
  __shared__ __align__(16) _Float16 Ks[64 * 72];      // [key][d]
  __shared__ __align__(16) _Float16 Vt[64 * 72];      // [d][key]
  __shared__ __align__(16) _Float16 Ps[4][16 * 72];   // per-wave P tile

  const int t = threadIdx.x;
  const int lane = t & 31;
  const int wv = t >> 5;                 // 0..3
  const int h = (lane >> 4) & 1;
  const int qb = blockIdx.x;
  const int bh = blockIdx.y;
  const size_t base = (size_t)bh * SEQ * HD;
  const int s0 = qb * 64;
  const float scale = 0.125f;            // 1/sqrt(64)

  for (int i = t; i < 64 * 64; i += 128) {
    int r = i >> 6, c = i & 63;
    Qs[r * 72 + c] = Q[base + (size_t)(s0 + r) * HD + c];
  }
  __syncthreads();
  v16h qf[2];
  qf[0] = afrag_lds(Qs + (wv * 16) * 72, 72, lane & 15, 0, lane);
  qf[1] = afrag_lds(Qs + (wv * 16) * 72, 72, lane & 15, 32, lane);

  float mrow[8], lrow[8];
  v8f acc[4] = {};
#pragma unroll
  for (int r = 0; r < 8; ++r) { mrow[r] = -1e30f; lrow[r] = 0.f; }

  for (int kb = 0; kb < SEQ; kb += 64) {
    __syncthreads();
    for (int i = t; i < 64 * 64; i += 128) {
      int r = i >> 6, c = i & 63;
      _Float16 kv = Kg[base + (size_t)(kb + r) * HD + c];
      _Float16 vv = V[base + (size_t)(kb + r) * HD + c];
      Ks[r * 72 + c] = kv;
      Vt[c * 72 + r] = vv;               // store V transposed
    }
    if (kb + 64 < SEQ) {                 // prefetch next KV tile
      __builtin_prefetch(&Kg[base + (size_t)(kb + 64 + (t >> 1)) * HD], 0, 1);
      __builtin_prefetch(&V[base + (size_t)(kb + 64 + (t >> 1)) * HD], 0, 1);
    }
    __syncthreads();

    // scores: 16x64 = 4 n-tiles, K=64 in two 32-chunks
    v8f sc[4];
#pragma unroll
    for (int j = 0; j < 4; ++j) {
      v8f s = {};
#pragma unroll
      for (int kc = 0; kc < 2; ++kc) {
        v16h kf = bfrag_lds(Ks + (j * 16) * 72, 72, lane & 15, kc * 32, lane);
        s = wmma16(qf[kc], kf, s);
      }
#pragma unroll
      for (int r = 0; r < 8; ++r) s[r] *= scale;
      sc[j] = s;
    }

    // online softmax (rows live across the 16 lanes sharing h)
#pragma unroll
    for (int r = 0; r < 8; ++r) {
      float mx = fmaxf(fmaxf(sc[0][r], sc[1][r]), fmaxf(sc[2][r], sc[3][r]));
#pragma unroll
      for (int off = 1; off < 16; off <<= 1)
        mx = fmaxf(mx, __shfl_xor(mx, off, 32));
      const float mnew = fmaxf(mrow[r], mx);
      const float alpha = __expf(mrow[r] - mnew);
      mrow[r] = mnew;
      float rs = 0.f;
#pragma unroll
      for (int j = 0; j < 4; ++j) {
        float p = __expf(sc[j][r] - mnew);
        sc[j][r] = p;
        rs += p;
      }
#pragma unroll
      for (int off = 1; off < 16; off <<= 1)
        rs += __shfl_xor(rs, off, 32);
      lrow[r] = lrow[r] * alpha + rs;
#pragma unroll
      for (int j = 0; j < 4; ++j) acc[j][r] *= alpha;
    }

    // transpose P through per-wave LDS into A-fragment layout
    _Float16* pw = &Ps[wv][0];
#pragma unroll
    for (int j = 0; j < 4; ++j)
#pragma unroll
      for (int r = 0; r < 8; ++r)
        pw[(r + 8 * h) * 72 + j * 16 + (lane & 15)] = (_Float16)sc[j][r];
    __syncthreads();

    v16h pf[2];
    pf[0] = afrag_lds(pw, 72, lane & 15, 0, lane);
    pf[1] = afrag_lds(pw, 72, lane & 15, 32, lane);
#pragma unroll
    for (int jd = 0; jd < 4; ++jd) {
#pragma unroll
      for (int kc = 0; kc < 2; ++kc) {
        v16h vf = bfrag_lds(Vt + (jd * 16) * 72, 72, lane & 15, kc * 32, lane);
        acc[jd] = wmma16(pf[kc], vf, acc[jd]);
      }
    }
  }

  const int b = bh / HEADS, head = bh % HEADS;
#pragma unroll
  for (int jd = 0; jd < 4; ++jd) {
    const int d = head * HD + jd * 16 + (lane & 15);
#pragma unroll
    for (int r = 0; r < 8; ++r) {
      const int gm = b * SEQ + s0 + wv * 16 + r + 8 * h;
      ctx[(size_t)gm * INNER + d] = (_Float16)(acc[jd][r] / lrow[r]);
    }
  }
}

// ---------- Kernel 3: output projection  (ctx[4096x768] @ w[768x768] + b) ----------
// Same 128x128 block / 32x64 wave tiling as kernel 1; A already f16.

__global__ __launch_bounds__(256) void outproj_kernel(
    const _Float16* __restrict__ ctx, const float* __restrict__ w,
    const float* __restrict__ bias, float* __restrict__ out) {
  __shared__ __align__(16) _Float16 As[128 * 72];
  __shared__ __align__(16) _Float16 Bs[128 * 72];

  const int t = threadIdx.x;
  const int lane = t & 31;
  const int wv = t >> 5;
  const int wm = wv & 3;
  const int wn = wv >> 2;
  const int m0 = blockIdx.x * 128;
  const int n0 = blockIdx.y * 128;
  const int h = (lane >> 4) & 1;

  v8f acc[2][4] = {};
  for (int kc = 0; kc < INNER; kc += 64) {
    for (int i = t; i < 128 * 64; i += 256) {
      int r = i >> 6, c = i & 63;
      As[r * 72 + c] = ctx[(size_t)(m0 + r) * INNER + kc + c];
    }
    for (int i = t; i < 64 * 128; i += 256) {
      int k = i >> 7, n = i & 127;
      Bs[n * 72 + k] = (_Float16)w[(size_t)(kc + k) * DIM + n0 + n];
    }
    if (kc + 64 < INNER) {
      __builtin_prefetch(&ctx[(size_t)(m0 + (t >> 1)) * INNER + kc + 64], 0, 1);
      __builtin_prefetch(&w[(size_t)(kc + 64 + (t >> 2)) * DIM + n0 + (t & 3) * 32], 0, 1);
    }
    __syncthreads();
#pragma unroll
    for (int kk = 0; kk < 2; ++kk) {
      v16h a0 = afrag_lds(As + (wm * 32) * 72, 72, lane & 15, kk * 32, lane);
      v16h a1 = afrag_lds(As + (wm * 32 + 16) * 72, 72, lane & 15, kk * 32, lane);
#pragma unroll
      for (int j = 0; j < 4; ++j) {
        v16h b = bfrag_lds(Bs + (wn * 64 + j * 16) * 72, 72, lane & 15,
                           kk * 32, lane);
        acc[0][j] = wmma16(a0, b, acc[0][j]);
        acc[1][j] = wmma16(a1, b, acc[1][j]);
      }
    }
    __syncthreads();
  }

#pragma unroll
  for (int ms = 0; ms < 2; ++ms)
#pragma unroll
    for (int j = 0; j < 4; ++j) {
      const int gn = n0 + wn * 64 + j * 16 + (lane & 15);
      const float bv = bias[gn];
#pragma unroll
      for (int r = 0; r < 8; ++r) {
        const int gm = m0 + wm * 32 + ms * 16 + r + 8 * h;
        out[(size_t)gm * DIM + gn] = acc[ms][j][r] + bv;
      }
    }
}

// ---------- launch ----------

extern "C" void kernel_launch(void* const* d_in, const int* in_sizes, int n_in,
                              void* d_out, int out_size, void* d_ws,
                              size_t ws_size, hipStream_t stream) {
  const float* x     = (const float*)d_in[0];
  const float* w_qkv = (const float*)d_in[1];
  const float* b_qkv = (const float*)d_in[2];
  const float* w_out = (const float*)d_in[3];
  const float* b_out = (const float*)d_in[4];
  float* out = (float*)d_out;

  const size_t per = (size_t)BATCH * HEADS * SEQ * HD;   // 3,145,728 f16
  _Float16* Q   = (_Float16*)d_ws;
  _Float16* K   = Q + per;
  _Float16* V   = K + per;
  _Float16* ctx = V + per;

  qkv_kernel<<<dim3(ROWS / 128, NQKV / 128), 256, 0, stream>>>(x, w_qkv, b_qkv,
                                                               Q, K, V);
  attn_kernel<<<dim3(SEQ / 64, BATCH * HEADS), 128, 0, stream>>>(Q, K, V, ctx);
  outproj_kernel<<<dim3(ROWS / 128, DIM / 128), 256, 0, stream>>>(ctx, w_out,
                                                                  b_out, out);
}
